// Mamba_53936199303795
// MI455X (gfx1250) — matrix operand order repaired and checked by reading, not compiled, not run on hardware
//
#include <hip/hip_runtime.h>

typedef __attribute__((ext_vector_type(16))) __bf16 v16bf;
typedef __attribute__((ext_vector_type(8)))  float  v8f;
typedef __attribute__((ext_vector_type(4)))  unsigned int u32x4;
typedef __attribute__((ext_vector_type(8)))  int i32x8;
typedef __attribute__((ext_vector_type(4)))  int i32x4;

#define B_  64
#define T_  1024
#define IN_ 256
#define H_  512
#define S_  512
#define C_  10

__device__ inline unsigned short f2bf(float f) {
  unsigned int u = __float_as_uint(f);
  unsigned int r = (u + 0x7FFFu + ((u >> 16) & 1u)) >> 16;   // RNE
  return (unsigned short)r;
}
__device__ inline float bf2f(unsigned short h) {
  return __uint_as_float(((unsigned int)h) << 16);
}

// LDS byte offset of a shared-memory pointer (DS-style address, what the TDM
// descriptor's lds_addr field wants): addrspacecast generic->local, ptrtoint.
typedef __attribute__((address_space(3))) unsigned short lds_us;
__device__ inline unsigned lds_addr_of(unsigned short* p) {
  return (unsigned)(unsigned long long)(lds_us*)p;
}

union FragU { v16bf v; uint4 q[2]; };

// A-fragment (16x32 bf16, row-major source, per-lane row index):
// lane half 0 holds K in {k0+0..7, k0+16..23}; half 1 holds {k0+8..15, k0+24..31}
__device__ inline v16bf load_a_frag(const unsigned short* p, int row, int stride,
                                    int k0, int half) {
  FragU u;
  const unsigned short* base = p + row * stride + k0 + half * 8;
  u.q[0] = *(const uint4*)(base);
  u.q[1] = *(const uint4*)(base + 16);
  return u.v;
}
// B-fragment (32x16 bf16). Row-major source holds one matrix column per row
// (i.e. Bg[s, h]): lane's column = lane&15, half 0 holds K in {k0+0..7, k0+16..23}; half 1 holds {k0+8..15, k0+24..31}.
__device__ inline v16bf load_b_frag(const unsigned short* p, int row, int stride,
                                    int k0, int half) {
  FragU u;
  const unsigned short* base = p + row * stride + k0 + half * 8;
  u.q[0] = *(const uint4*)(base);
  u.q[1] = *(const uint4*)(base + 16);
  return u.v;
}

// Issue a TDM load of a [rows=16 x 512 bf16] row-major tile into LDS.
// Descriptor layout per CDNA5 ISA (2D: groups 2/3 zero).
__device__ inline void tdm_load_tile(const unsigned short* gsrc, unsigned lds_byte_addr) {
  unsigned long long ga = (unsigned long long)gsrc;
  u32x4 g0;
  g0.x = 1u;                                        // count=1, user mode
  g0.y = lds_byte_addr;                             // lds_addr [63:32]
  g0.z = (unsigned)ga;                              // global_addr [95:64]
  g0.w = (unsigned)(ga >> 32) | (2u << 30);         // addr[56:32] | type=2
  i32x8 g1;
  g1[0] = 0x00010000;                               // data_size=1 (2 bytes)
  g1[1] = (int)(512u << 16);                        // tensor_dim0 = 512 (lo16)
  g1[2] = (int)(16u << 16);                         // tensor_dim1 = 16
  g1[3] = (int)(512u << 16);                        // tile_dim0 = 512
  g1[4] = 16;                                       // tile_dim1 = 16
  g1[5] = 512;                                      // tensor_dim0_stride = 512
  g1[6] = 0;
  g1[7] = 0;
  i32x4 z4 = {0, 0, 0, 0};
#if __clang_major__ >= 23
  i32x8 z8 = {0, 0, 0, 0, 0, 0, 0, 0};
  __builtin_amdgcn_tensor_load_to_lds(g0, g1, z4, z4, z8, 0);
#else
  __builtin_amdgcn_tensor_load_to_lds(g0, g1, z4, z4, 0);
#endif
}

// ---------------------------------------------------------------- prep ------
__global__ void prep_weights(const float* __restrict__ Wi, const float* __restrict__ Bg,
                             unsigned short* __restrict__ Wi16,
                             unsigned short* __restrict__ Bg16) {
  int i = blockIdx.x * blockDim.x + threadIdx.x;
  if (i < H_ * IN_) Wi16[i] = f2bf(Wi[i]);
  if (i < S_ * H_)  Bg16[i] = f2bf(Bg[i]);
}

// ------------------------------------------------------------ gate GEMM -----
// gate[b,t,h] = silu(x[b,t,:] . W_in[h,:] + Wb[h]) stored as bf16.
// One WG = 16 rows of (B*T), all 512 H columns; 8 waves x 4 N-tiles.
__global__ __launch_bounds__(256) void gate_kernel(const float* __restrict__ x,
                                                   const unsigned short* __restrict__ Wi16,
                                                   const float* __restrict__ Wb,
                                                   unsigned short* __restrict__ gate) {
  __shared__ __align__(16) unsigned short xt[16 * IN_];   // 8KB x tile as bf16
  int tid  = threadIdx.x;
  int row0 = blockIdx.x * 16;
  const float* xsrc = x + (size_t)row0 * IN_;
#pragma unroll
  for (int i = 0; i < 16; ++i) {
    int e = i * 256 + tid;            // coalesced
    xt[e] = f2bf(xsrc[e]);
  }
  __syncthreads();

  int lane = tid & 31, wave = tid >> 5, half = lane >> 4, l15 = lane & 15;

  v8f acc0 = {0,0,0,0,0,0,0,0};
  v8f acc1 = acc0, acc2 = acc0, acc3 = acc0;

#pragma unroll
  for (int k = 0; k < IN_ / 32; ++k) {
    v16bf a = load_a_frag(xt, l15, IN_, k * 32, half);
    int ncol0 = wave * 64 + l15;
    v16bf b0 = load_b_frag(Wi16, ncol0 +  0, IN_, k * 32, half);
    v16bf b1 = load_b_frag(Wi16, ncol0 + 16, IN_, k * 32, half);
    v16bf b2 = load_b_frag(Wi16, ncol0 + 32, IN_, k * 32, half);
    v16bf b3 = load_b_frag(Wi16, ncol0 + 48, IN_, k * 32, half);
    acc0 = __builtin_amdgcn_wmma_f32_16x16x32_bf16(false, a, false, b0, (short)0, acc0, false, false);
    acc1 = __builtin_amdgcn_wmma_f32_16x16x32_bf16(false, a, false, b1, (short)0, acc1, false, false);
    acc2 = __builtin_amdgcn_wmma_f32_16x16x32_bf16(false, a, false, b2, (short)0, acc2, false, false);
    acc3 = __builtin_amdgcn_wmma_f32_16x16x32_bf16(false, a, false, b3, (short)0, acc3, false, false);
  }

  v8f accs[4] = {acc0, acc1, acc2, acc3};
#pragma unroll
  for (int nt = 0; nt < 4; ++nt) {
    int ncol = wave * 64 + nt * 16 + l15;
    float bias = Wb[ncol];
#pragma unroll
    for (int j = 0; j < 8; ++j) {
      int m = half * 8 + j;                        // C layout: VGPR j -> M=j / M=j+8
      float v = accs[nt][j] + bias;
      v = v / (1.0f + __expf(-v));                 // silu
      gate[(size_t)(row0 + m) * H_ + ncol] = f2bf(v);
    }
  }
}

// ----------------------------------------------------- weighted scan --------
// s_last[b,s] = sum_t d[s]^(T-1-t) * (gate[b,t,:] . Bg[s,:])
// grid (4 s-groups, 64 b); 8 waves x 16 S columns. Bg fragments live in VGPRs.
// Gate tiles are streamed into double-buffered LDS by the Tensor Data Mover
// (wave 0 issues TDM + s_wait_tensorcnt; workgroup barrier publishes to all).
__global__ __launch_bounds__(256) void scan_kernel(const unsigned short* __restrict__ gate,
                                                   const unsigned short* __restrict__ Bg16,
                                                   const float* __restrict__ decay,
                                                   float* __restrict__ s_last) {
  __shared__ __align__(16) unsigned short gt[2][16 * H_];   // 2 x 16KB gate tiles
  int tid = threadIdx.x, lane = tid & 31, wave = tid >> 5;
  int half = lane >> 4, l15 = lane & 15;
  int b    = blockIdx.y;
  int scol = blockIdx.x * 128 + wave * 16 + l15;

  float dd  = 1.0f / (1.0f + __expf(-decay[scol]));         // sigmoid(decay)
  float d2 = dd * dd, d4 = d2 * d2, d8 = d4 * d4, d16 = d8 * d8;

  v16bf Bf[16];                                             // 128 VGPRs of Bg
#pragma unroll
  for (int k = 0; k < 16; ++k) Bf[k] = load_b_frag(Bg16, scol, H_, k * 32, half);

  float state = 0.0f;
  const unsigned short* gb = gate + (size_t)b * T_ * H_;
  unsigned lds0 = lds_addr_of(&gt[0][0]);
  unsigned lds1 = lds_addr_of(&gt[1][0]);

  if (wave == 0) tdm_load_tile(gb, lds0);                   // prologue: tile 0

  for (int tt = 0; tt < T_ / 16; ++tt) {
    if (wave == 0) {
      if (tt + 1 < T_ / 16) {
        tdm_load_tile(gb + (size_t)(tt + 1) * 16 * H_, ((tt + 1) & 1) ? lds1 : lds0);
        __builtin_amdgcn_s_wait_tensorcnt(1);   // tile tt landed (in-order TDM)
      } else {
        __builtin_amdgcn_s_wait_tensorcnt(0);
      }
    }
    __syncthreads();                                        // tile tt visible to all

    const unsigned short* cur = gt[tt & 1];
    v8f acc = {0,0,0,0,0,0,0,0};
#pragma unroll
    for (int k = 0; k < 16; ++k) {
      v16bf a = load_a_frag(cur, l15, H_, k * 32, half);
      acc = __builtin_amdgcn_wmma_f32_16x16x32_bf16(false, a, false, Bf[k], (short)0, acc, false, false);
    }
    // Horner over this lane-half's 8 rows: part = sum_j d^(7-j) * u_j
    float part = 0.0f;
#pragma unroll
    for (int j = 0; j < 8; ++j) part = part * dd + acc[j];
    // combine halves (lanes 0-15 hold rows 0-7, lanes 16-31 hold rows 8-15)
    float other = __shfl_xor(part, 16, 32);
    float total = (half == 0) ? (d8 * part + other) : (d8 * other + part);
    state = state * d16 + total;
    __syncthreads();    // readers done before TDM overwrites this buffer
  }
  if (half == 0) s_last[(size_t)b * S_ + scol] = state;
}

// --------------------------------------------------------------- epilogue ---
__global__ __launch_bounds__(512) void final_kernel(const float* __restrict__ x,
                                                    const unsigned short* __restrict__ gate,
                                                    const float* __restrict__ Ch,
                                                    const float* __restrict__ Dw,
                                                    const float* __restrict__ Db,
                                                    const float* __restrict__ fcw,
                                                    const float* __restrict__ fcb,
                                                    const float* __restrict__ s_last,
                                                    float* __restrict__ out) {
  __shared__ float hbuf[H_];
  int b = blockIdx.x, h = threadIdx.x;
  const float* xl = x + ((size_t)b * T_ + (T_ - 1)) * IN_;
  float g = bf2f(gate[((size_t)b * T_ + (T_ - 1)) * H_ + h]);
  float acc = Db[h];
  const float* sl = s_last + (size_t)b * S_;
  const float* ch = Ch + (size_t)h * S_;
  for (int s = 0; s < S_; ++s) acc += sl[s] * ch[s];
  const float* dw = Dw + (size_t)h * IN_;
  for (int i = 0; i < IN_; ++i) acc += xl[i] * dw[i];
  hbuf[h] = g * acc;
  __syncthreads();
  if (h < C_) {
    float o = fcb[h];
    const float* fw = fcw + (size_t)h * H_;
    for (int k = 0; k < H_; ++k) o += hbuf[k] * fw[k];
    out[b * C_ + h] = o;
  }
}

// ---------------------------------------------------------------------------
extern "C" void kernel_launch(void* const* d_in, const int* in_sizes, int n_in,
                              void* d_out, int out_size, void* d_ws, size_t ws_size,
                              hipStream_t stream) {
  const float* x     = (const float*)d_in[0];
  const float* Wi    = (const float*)d_in[1];
  const float* Wb    = (const float*)d_in[2];
  const float* decay = (const float*)d_in[3];
  const float* Bg    = (const float*)d_in[4];
  const float* Ch    = (const float*)d_in[5];
  const float* Dw    = (const float*)d_in[6];
  const float* Db    = (const float*)d_in[7];
  const float* fcw   = (const float*)d_in[8];
  const float* fcb   = (const float*)d_in[9];
  float* out = (float*)d_out;

  char* ws = (char*)d_ws;
  size_t off = 0;
  unsigned short* Wi16 = (unsigned short*)(ws + off); off += (size_t)H_ * IN_ * 2;     // 256KB
  unsigned short* Bg16 = (unsigned short*)(ws + off); off += (size_t)S_ * H_ * 2;      // 512KB
  unsigned short* gate = (unsigned short*)(ws + off); off += (size_t)B_ * T_ * H_ * 2; // 64MB
  float* s_last        = (float*)(ws + off);                                           // 128KB

  prep_weights<<<(S_ * H_ + 255) / 256, 256, 0, stream>>>(Wi, Bg, Wi16, Bg16);
  gate_kernel<<<B_ * T_ / 16, 256, 0, stream>>>(x, Wi16, Wb, gate);
  dim3 sg(S_ / 128, B_);
  scan_kernel<<<sg, 256, 0, stream>>>(gate, Bg16, decay, s_last);
  final_kernel<<<B_, 512, 0, stream>>>(x, gate, Ch, Dw, Db, fcw, fcb, s_last, out);
}
